// AligningModel_37245956391160
// MI455X (gfx1250) — matrix operand order repaired and checked by reading, not compiled, not run on hardware
//
#include <hip/hip_runtime.h>

typedef _Float16 f16_t;
typedef f16_t v16h __attribute__((ext_vector_type(16)));
typedef f16_t v8h  __attribute__((ext_vector_type(8)));
typedef f16_t v4h  __attribute__((ext_vector_type(4)));
typedef float v8f  __attribute__((ext_vector_type(8)));
typedef float v4f  __attribute__((ext_vector_type(4)));

#define WAVE 32

// ---------------------------------------------------------------------------
// CDNA5 WMMA fragment builders (wave32; 16-bit A 16x32, B 32x16, f32 C 16x16)
// A element j -> K = (j<8 ? j+8*hi : j+8+8*hi): two contiguous 8-elem runs.
// B element j -> K = (j<8 ? j+8*hi : j+8+8*hi): same as A.
// ---------------------------------------------------------------------------
__device__ __forceinline__ v16h a_frag_f32(const float* rc, int hi) {
    const v4f* p0 = (const v4f*)(rc + 8 * hi);
    const v4f* p1 = (const v4f*)(rc + 16 + 8 * hi);
    v4f x0 = p0[0], x1 = p0[1], x2 = p1[0], x3 = p1[1];
    v16h a;
    #pragma unroll
    for (int i = 0; i < 4; ++i) {
        a[i]      = (f16_t)x0[i];
        a[4 + i]  = (f16_t)x1[i];
        a[8 + i]  = (f16_t)x2[i];
        a[12 + i] = (f16_t)x3[i];
    }
    return a;
}
// f16 source (LDS-staged window): exactly two b128 loads, no converts
__device__ __forceinline__ v16h a_frag_f16(const f16_t* rc, int hi) {
    const v8h* p0 = (const v8h*)(rc + 8 * hi);
    const v8h* p1 = (const v8h*)(rc + 16 + 8 * hi);
    v8h a0 = p0[0], a1 = p1[0];
    v16h a;
    #pragma unroll
    for (int i = 0; i < 8; ++i) { a[i] = a0[i]; a[8 + i] = a1[i]; }
    return a;
}
__device__ __forceinline__ v16h b_frag_f16(const f16_t* cc, int hi) {
    const v8h* p0 = (const v8h*)(cc + 8 * hi);
    const v8h* p1 = (const v8h*)(cc + 16 + 8 * hi);
    v8h b0 = p0[0], b1 = p1[0];
    v16h b;
    #pragma unroll
    for (int i = 0; i < 8; ++i) { b[i] = b0[i]; b[8 + i] = b1[i]; }
    return b;
}
__device__ __forceinline__ float sigmoidf_(float x) {
    return 1.0f / (1.0f + __expf(-x));
}

// ---------------------------------------------------------------------------
// GLU conv block with LDS-staged activation window.
// One block = (batch, 16-row t-tile). The 18-row x C conv window is staged to
// LDS once (masked + f32->f16), then the 8 waves sweep all C/16 n-tiles with
// two WMMA chains each (a-half / g-half of the 2C conv output).
// Wt is f16 transposed: Wt[n][3C], n in [0,2C).
// ---------------------------------------------------------------------------
__global__ __launch_bounds__(256)
void glu_conv_block(const float* __restrict__ x, float* __restrict__ out,
                    const f16_t* __restrict__ Wt, const float* __restrict__ bias,
                    const int* __restrict__ lens, int maskGT,
                    const float* __restrict__ zrow,
                    int Tpad, int Tlen, int C, int ld, int tilesT)
{
    __shared__ __align__(16) f16_t xs[18 * 768];

    int b   = blockIdx.x / tilesT;
    int t0  = (blockIdx.x % tilesT) * 16;
    int tid = (int)threadIdx.x;
    int wv  = tid / WAVE, lane = tid & 31, lo = lane & 15, hi = lane >> 4;
    int len = lens[b];

    const float* xb = x   + (size_t)b * Tpad * ld;
    float*       ob = out + (size_t)b * Tpad * ld;

    // ---- stage rows [t0-1, t0+16] (masked, converted) into LDS ----
    {
        int cc = tid * 4;
        #pragma unroll
        for (int row = 0; row < 18; ++row) {
            int tt = t0 - 1 + row;
            bool vr = (tt >= 0) && (tt < Tlen) && (maskGT ? (tt <= len) : (tt < len));
            const float* xr = vr ? (xb + (ptrdiff_t)tt * ld) : zrow;
            if (cc < C) {
                v4f xv = *(const v4f*)(xr + cc);
                v4h hv;
                #pragma unroll
                for (int i = 0; i < 4; ++i) hv[i] = (f16_t)xv[i];
                *(v4h*)(xs + row * C + cc) = hv;
            }
        }
    }
    __syncthreads();

    const int K3 = 3 * C;
    int nTt = C / 16;
    for (int nt = wv; nt < nTt; nt += 8) {
        int colA = nt * 16 + lo;
        const f16_t* wtA = Wt + (size_t)colA * K3;
        const f16_t* wtG = Wt + (size_t)(C + colA) * K3;

        v8f accA = {0.f,0.f,0.f,0.f,0.f,0.f,0.f,0.f};
        v8f accG = {0.f,0.f,0.f,0.f,0.f,0.f,0.f,0.f};

        #pragma unroll
        for (int ks = 0; ks < 3; ++ks) {
            const f16_t* ar = xs + (lo + ks) * C;   // local row of the staged window
            const f16_t* wA = wtA + ks * C;
            const f16_t* wG = wtG + ks * C;
            for (int c = 0; c < C; c += 32) {
                v16h af = a_frag_f16(ar + c, hi);
                v16h bA = b_frag_f16(wA + c, hi);
                v16h bG = b_frag_f16(wG + c, hi);
                accA = __builtin_amdgcn_wmma_f32_16x16x32_f16(false, af, false, bA, (short)0, accA, false, false);
                accG = __builtin_amdgcn_wmma_f32_16x16x32_f16(false, af, false, bG, (short)0, accG, false, false);
            }
        }

        float ba = bias[colA];
        float bg = bias[C + colA];
        #pragma unroll
        for (int i = 0; i < 8; ++i) {
            int t = t0 + i + 8 * hi;
            bool vr = (t < Tlen) && (maskGT ? (t <= len) : (t < len));
            const float* rr = vr ? (xb + (size_t)t * ld) : zrow;
            float xres = rr[colA];
            float a = accA[i] + ba;
            float g = accG[i] + bg;
            ob[(size_t)t * ld + colA] = a * sigmoidf_(g) + xres;
        }
    }
}

// ---------------------------------------------------------------------------
// GEMM + bias (+tanh): out[M,N] = act(X[M,Kp] @ Wt^T + b); Kp % 32 == 0,
// Wt packed f16 [N][Kp]. One wave per 16x16 tile.
// ---------------------------------------------------------------------------
__global__ __launch_bounds__(256)
void gemm_bias(const float* __restrict__ X, int ldx,
               const f16_t* __restrict__ Wt, const float* __restrict__ bias,
               float* __restrict__ out, int ldo,
               int N, int K, int totalTiles, int act)
{
    int wv = blockIdx.x * (blockDim.x / WAVE) + ((int)threadIdx.x / WAVE);
    if (wv >= totalTiles) return;
    int lane = (int)threadIdx.x & 31, lo = lane & 15, hi = lane >> 4;
    int nT = N / 16;
    int m0 = (wv / nT) * 16;
    int n0 = (wv % nT) * 16;
    int col = n0 + lo;

    const float* xr = X + (size_t)(m0 + lo) * ldx;
    const f16_t* wc = Wt + (size_t)col * K;
    __builtin_prefetch(xr, 0, 0);

    v8f acc = {0.f,0.f,0.f,0.f,0.f,0.f,0.f,0.f};
    for (int kc = 0; kc < K; kc += 32) {
        v16h af = a_frag_f32(xr + kc, hi);
        v16h bf = b_frag_f16(wc + kc, hi);
        acc = __builtin_amdgcn_wmma_f32_16x16x32_f16(false, af, false, bf, (short)0, acc, false, false);
    }

    float bb = bias[col];
    #pragma unroll
    for (int i = 0; i < 8; ++i) {
        float v = acc[i] + bb;
        if (act == 1) v = tanhf(v);
        out[(size_t)(m0 + i + 8 * hi) * ldo + col] = v;
    }
}

// ---------------------------------------------------------------------------
// Fused attention. Scores via WMMA (mel f32 A, ph16 f16 B), masked softmax in
// LDS, context via WMMA (LDS f32 A, phT16 f16 B with S padded to 288 zeros).
// Uses the softmax-invariant score 2*m.p - |p|^2; masked/pad -> -1e9.
// ---------------------------------------------------------------------------
__global__ __launch_bounds__(256)
void attn_context(const float* __restrict__ mel, int ldm,
                  const f16_t* __restrict__ ph16,   // [B][272][384]
                  const f16_t* __restrict__ phT16,  // [B][384][288]
                  const float* __restrict__ pnorm,  // [B][288]
                  const int* __restrict__ ph_lens,
                  float* __restrict__ ctx,
                  int T, int Spad /*272*/, int SpadT /*288*/, int Slen /*257*/, int E /*384*/)
{
    __shared__ __align__(16) float S[16][296];

    int tilesT = T / 16;
    int b  = blockIdx.x / tilesT;
    int t0 = (blockIdx.x % tilesT) * 16;
    int tid = (int)threadIdx.x;
    int wv = tid / WAVE, lane = tid & 31, lo = lane & 15, hi = lane >> 4;
    int plen = ph_lens[b];

    const float* melb = mel   + (size_t)b * T * ldm;
    const f16_t* phb  = ph16  + (size_t)b * Spad * E;
    const f16_t* phTb = phT16 + (size_t)b * E * SpadT;
    const float* pnb  = pnorm + (size_t)b * SpadT;

    S[tid / 16][272 + (tid & 15)] = 0.f;   // zero pad cols [272,288)
    __syncthreads();

    // scores
    int sTiles = Spad / 16;   // 17
    for (int st = wv; st < sTiles; st += 8) {
        int s0 = st * 16;
        v8f acc = {0.f,0.f,0.f,0.f,0.f,0.f,0.f,0.f};
        const float* mr = melb + (size_t)(t0 + lo) * ldm;
        const f16_t* pr = phb + (size_t)(s0 + lo) * E;
        for (int kc = 0; kc < E; kc += 32) {
            v16h af = a_frag_f32(mr + kc, hi);
            v16h bf = b_frag_f16(pr + kc, hi);
            acc = __builtin_amdgcn_wmma_f32_16x16x32_f16(false, af, false, bf, (short)0, acc, false, false);
        }
        int s = s0 + lo;
        bool ok = (s < Slen) && (s <= plen);   // ph_mask uses strict '>'
        float p2 = pnb[s];
        #pragma unroll
        for (int i = 0; i < 8; ++i)
            S[i + 8 * hi][s] = ok ? (2.0f * acc[i] - p2) : -1e9f;
    }
    __syncthreads();

    // row softmax over 272
    for (int r = wv; r < 16; r += 8) {
        float m = -3.4e38f;
        for (int s = lane; s < 272; s += 32) m = fmaxf(m, S[r][s]);
        #pragma unroll
        for (int off = 16; off >= 1; off >>= 1) m = fmaxf(m, __shfl_xor(m, off, 32));
        float sum = 0.f;
        for (int s = lane; s < 272; s += 32) sum += __expf(S[r][s] - m);
        #pragma unroll
        for (int off = 16; off >= 1; off >>= 1) sum += __shfl_xor(sum, off, 32);
        float inv = 1.0f / sum;
        for (int s = lane; s < 272; s += 32) S[r][s] = __expf(S[r][s] - m) * inv;
    }
    __syncthreads();

    // context = attn @ ph_enc
    int nE = E / 16;          // 24
    float* cb = ctx + (size_t)b * T * ldm;
    for (int nt = wv; nt < nE; nt += 8) {
        int n0 = nt * 16;
        v8f acc = {0.f,0.f,0.f,0.f,0.f,0.f,0.f,0.f};
        const f16_t* pc = phTb + (size_t)(n0 + lo) * SpadT;
        for (int kc = 0; kc < SpadT; kc += 32) {
            v16h af = a_frag_f32(&S[lo][kc], hi);
            v16h bf = b_frag_f16(pc + kc, hi);
            acc = __builtin_amdgcn_wmma_f32_16x16x32_f16(false, af, false, bf, (short)0, acc, false, false);
        }
        #pragma unroll
        for (int i = 0; i < 8; ++i)
            cb[(size_t)(t0 + i + 8 * hi) * ldm + n0 + lo] = acc[i];
    }
}

// ---------------------------------------------------------------------------
// log_softmax over V=256, one wave per row, writes cols [0,256) of output
// ---------------------------------------------------------------------------
__global__ __launch_bounds__(256)
void log_softmax_k(const float* __restrict__ logits, float* __restrict__ out,
                   int rows, int ldo)
{
    int wv = blockIdx.x * 8 + (int)threadIdx.x / WAVE;
    if (wv >= rows) return;
    int lane = (int)threadIdx.x & 31;
    const float* lr = logits + (size_t)wv * 256;
    float v[8];
    float m = -3.4e38f;
    #pragma unroll
    for (int q = 0; q < 8; ++q) { v[q] = lr[lane + 32 * q]; m = fmaxf(m, v[q]); }
    #pragma unroll
    for (int off = 16; off >= 1; off >>= 1) m = fmaxf(m, __shfl_xor(m, off, 32));
    float sum = 0.f;
    #pragma unroll
    for (int q = 0; q < 8; ++q) sum += __expf(v[q] - m);
    #pragma unroll
    for (int off = 16; off >= 1; off >>= 1) sum += __shfl_xor(sum, off, 32);
    float lz = __logf(sum);
    float* orow = out + (size_t)wv * ldo;
    #pragma unroll
    for (int q = 0; q < 8; ++q) orow[lane + 32 * q] = v[q] - m - lz;
}

// ---------------------------------------------------------------------------
// small prep kernels
// ---------------------------------------------------------------------------
__global__ __launch_bounds__(256)
void embed_k(const int* __restrict__ phon, const float* __restrict__ emb,
             float* __restrict__ out, int B_, int Sp, int E_, int Tph)
{
    int idx = blockIdx.x * blockDim.x + (int)threadIdx.x;
    int total = B_ * Sp * E_;
    if (idx >= total) return;
    int e = idx % E_;
    int s = (idx / E_) % Sp;
    int b = idx / (E_ * Sp);
    float val = 0.f;
    if (s < Tph + 1) {
        int tok = (s == 0) ? 0 : phon[b * Tph + (s - 1)];
        val = emb[(size_t)tok * E_ + e];
    }
    out[idx] = val;
}

// W [K,N] f32  ->  Wt [N,Kpad] f16 (zero-padded K)
__global__ __launch_bounds__(256)
void pack_wt(const float* __restrict__ W, f16_t* __restrict__ Wt, int K, int N, int Kpad)
{
    int idx = blockIdx.x * blockDim.x + (int)threadIdx.x;
    int total = N * Kpad;
    if (idx >= total) return;
    int k = idx % Kpad;
    int n = idx / Kpad;
    float v = (k < K) ? W[(size_t)k * N + n] : 0.f;
    Wt[idx] = (f16_t)v;
}

// X [rows,K] -> Y [rows,Kpad] zero-padded
__global__ __launch_bounds__(256)
void pad_rows(const float* __restrict__ X, float* __restrict__ Y, int rows, int K, int Kpad)
{
    int idx = blockIdx.x * blockDim.x + (int)threadIdx.x;
    int total = rows * Kpad;
    if (idx >= total) return;
    int k = idx % Kpad;
    int r = idx / Kpad;
    Y[idx] = (k < K) ? X[(size_t)r * K + k] : 0.f;
}

__global__ __launch_bounds__(256)
void fill_zero_f32(float* __restrict__ p, int n)
{
    int idx = blockIdx.x * blockDim.x + (int)threadIdx.x;
    if (idx < n) p[idx] = 0.f;
}

// ph_enc f32 [B][272][E] -> ph16 [B][272][E], phT16 [B][E][288] (zeros for
// s>=Slen)
__global__ __launch_bounds__(256)
void ph_finalize(const float* __restrict__ ph, f16_t* __restrict__ ph16,
                 f16_t* __restrict__ phT, int B_, int Sp, int SpT, int E_, int Slen)
{
    int idx = blockIdx.x * blockDim.x + (int)threadIdx.x;
    int total = B_ * SpT * E_;
    if (idx >= total) return;
    int e = idx % E_;
    int s = (idx / E_) % SpT;
    int b = idx / (E_ * SpT);
    float v = 0.f;
    if (s < Slen) v = ph[((size_t)b * Sp + s) * E_ + e];
    if (s < Sp) ph16[((size_t)b * Sp + s) * E_ + e] = (f16_t)v;
    phT[((size_t)b * E_ + e) * SpT + s] = (f16_t)v;
}

__global__ __launch_bounds__(256)
void ph_norms(const float* __restrict__ ph, float* __restrict__ pn,
              int B_, int Sp, int SpT, int E_, int Slen)
{
    int idx = blockIdx.x * blockDim.x + (int)threadIdx.x;
    if (idx >= B_ * SpT) return;
    int s = idx % SpT;
    int b = idx / SpT;
    float a = 0.f;
    if (s < Slen) {
        const float* pr = ph + ((size_t)b * Sp + s) * E_;
        for (int e = 0; e < E_; ++e) a += pr[e] * pr[e];
    }
    pn[idx] = a;
}

// ===========================================================================
extern "C" void kernel_launch(void* const* d_in, const int* in_sizes, int n_in,
                              void* d_out, int out_size, void* d_ws, size_t ws_size,
                              hipStream_t stream) {
    (void)in_sizes; (void)n_in; (void)out_size; (void)ws_size;

    const int B = 32, T = 2000, E = 384, V = 256, MEL = 80, DH = 128;
    const int SP = 272, SPT = 288, SL = 257;

    const float* mels         = (const float*)d_in[0];
    const int*   phonemes     = (const int*)  d_in[1];
    const int*   mel_lens     = (const int*)  d_in[2];
    const int*   phoneme_lens = (const int*)  d_in[3];
    const float* emb          = (const float*)d_in[4];
    const float* pe_W         = (const float*)d_in[5];
    const float* pe_b         = (const float*)d_in[6];
    const float* me_proj_W    = (const float*)d_in[7];
    const float* me_proj_b    = (const float*)d_in[8];
    const float* me_W         = (const float*)d_in[9];
    const float* me_b         = (const float*)d_in[10];
    const float* pd_W         = (const float*)d_in[11];
    const float* pd_b         = (const float*)d_in[12];
    const float* pd_lin_W     = (const float*)d_in[13];
    const float* pd_lin_b     = (const float*)d_in[14];
    const float* md_proj_W    = (const float*)d_in[15];
    const float* md_proj_b    = (const float*)d_in[16];
    const float* md_W         = (const float*)d_in[17];
    const float* md_b         = (const float*)d_in[18];
    const float* md_lin_W     = (const float*)d_in[19];
    const float* md_lin_b     = (const float*)d_in[20];
    float* out = (float*)d_out;

    // ---- workspace carve-up (f32 region then f16 region) ----
    float* buf0 = (float*)d_ws;                         // [B,T,768]
    size_t big = (size_t)B * T * 768;
    float* buf1   = buf0 + big;                         // [B,T,768]
    float* ph0    = buf1 + big;                         // [B,272,384]
    size_t phsz   = (size_t)B * SP * E;
    float* ph1    = ph0 + phsz;
    float* logits = ph1 + phsz;                         // [B,T,256]
    float* melpad = logits + (size_t)B * T * V;         // [B,T,96]
    float* zrow   = melpad + (size_t)B * T * 96;        // 1024 zeros
    float* pnorm  = zrow + 1024;                        // [B,288]
    f16_t* h = (f16_t*)(pnorm + (size_t)B * SPT);
    f16_t* ph16   = h;               h += (size_t)B * SP * E;        // [B,272,384]
    f16_t* phT16  = h;               h += (size_t)B * E * SPT;       // [B,384,288]
    f16_t* wt_me_proj = h;           h += (size_t)E * 96;
    f16_t* wt_pe  = h;               h += (size_t)4 * (2*E) * (3*E);
    f16_t* wt_me  = h;               h += (size_t)4 * (2*E) * (3*E);
    f16_t* wt_pd  = h;               h += (size_t)2 * 1536 * 2304;
    f16_t* wt_pd_lin = h;            h += (size_t)V * 768;
    f16_t* wt_md_proj = h;           h += (size_t)DH * V;
    f16_t* wt_md  = h;               h += (size_t)2 * (2*DH) * (3*DH);
    f16_t* wt_md_lin = h;            h += (size_t)MEL * DH;

    #define GRID1(total) (((total) + 255) / 256), 256, 0, stream

    // ---- one-time-per-launch prep: zero row, packed weights, padded mels ----
    fill_zero_f32<<<GRID1(1024)>>>(zrow, 1024);
    pack_wt<<<GRID1(E * 96)>>>(me_proj_W, wt_me_proj, MEL, E, 96);
    for (int i = 0; i < 4; ++i) {
        pack_wt<<<GRID1(2*E * 3*E)>>>(pe_W + (size_t)i*3*E*2*E, wt_pe + (size_t)i*(2*E)*(3*E), 3*E, 2*E, 3*E);
        pack_wt<<<GRID1(2*E * 3*E)>>>(me_W + (size_t)i*3*E*2*E, wt_me + (size_t)i*(2*E)*(3*E), 3*E, 2*E, 3*E);
    }
    for (int i = 0; i < 2; ++i) {
        pack_wt<<<GRID1(1536 * 2304)>>>(pd_W + (size_t)i*3*768*1536, wt_pd + (size_t)i*1536*2304, 2304, 1536, 2304);
        pack_wt<<<GRID1(2*DH * 3*DH)>>>(md_W + (size_t)i*3*DH*2*DH, wt_md + (size_t)i*(2*DH)*(3*DH), 3*DH, 2*DH, 3*DH);
    }
    pack_wt<<<GRID1(V * 768)>>>(pd_lin_W, wt_pd_lin, 768, V, 768);
    pack_wt<<<GRID1(DH * V)>>>(md_proj_W, wt_md_proj, V, DH, V);
    pack_wt<<<GRID1(MEL * DH)>>>(md_lin_W, wt_md_lin, DH, MEL, DH);
    pad_rows<<<GRID1(B * T * 96)>>>(mels, melpad, B * T, MEL, 96);

    // ---- phoneme branch: embed + 4 GLU blocks ----
    embed_k<<<GRID1(B * SP * E)>>>(phonemes, emb, ph0, B, SP, E, 256);
    {
        int tilesT = SP / 16;
        float* pin = ph0; float* pout = ph1;
        for (int i = 0; i < 4; ++i) {
            glu_conv_block<<<B * tilesT, 256, 0, stream>>>(
                pin, pout, wt_pe + (size_t)i*(2*E)*(3*E), pe_b + (size_t)i*2*E,
                phoneme_lens, 1, zrow, SP, SL, E, E, tilesT);
            float* tw = pin; pin = pout; pout = tw;
        }   // result in ph0
    }
    ph_finalize<<<GRID1(B * SPT * E)>>>(ph0, ph16, phT16, B, SP, SPT, E, SL);
    ph_norms<<<GRID1(B * SPT)>>>(ph0, pnorm, B, SP, SPT, E, SL);

    // ---- mel branch: 96->384 projection + 4 GLU blocks (left half of buf0) ----
    {
        int total = (B * T / 16) * (E / 16);
        gemm_bias<<<(total + 7) / 8, 256, 0, stream>>>(
            melpad, 96, wt_me_proj, me_proj_b, buf0, 768, E, 96, total, 0);
    }
    {
        int tilesT = T / 16;
        float* pin = buf0; float* pout = buf1;
        for (int i = 0; i < 4; ++i) {
            glu_conv_block<<<B * tilesT, 256, 0, stream>>>(
                pin, pout, wt_me + (size_t)i*(2*E)*(3*E), me_b + (size_t)i*2*E,
                mel_lens, 0, zrow, T, T, E, 768, tilesT);
            float* tw = pin; pin = pout; pout = tw;
        }   // mel_enc in buf0 cols [0,384)
    }

    // ---- fused attention: context into buf0 cols [384,768) ----
    attn_context<<<B * (T / 16), 256, 0, stream>>>(
        buf0, 768, ph16, phT16, pnorm, phoneme_lens, buf0 + E, T, SP, SPT, SL, E);

    // ---- decoder: 2 GLU blocks at C=768 ----
    {
        int tilesT = T / 16;
        float* pin = buf0; float* pout = buf1;
        for (int i = 0; i < 2; ++i) {
            glu_conv_block<<<B * tilesT, 256, 0, stream>>>(
                pin, pout, wt_pd + (size_t)i*1536*2304, pd_b + (size_t)i*1536,
                mel_lens, 0, zrow, T, T, 768, 768, tilesT);
            float* tw = pin; pin = pout; pout = tw;
        }   // dec in buf0
    }

    // ---- logits, log_softmax -> out[:, 0:256] ----
    {
        int total = (B * T / 16) * (V / 16);
        gemm_bias<<<(total + 7) / 8, 256, 0, stream>>>(
            buf0, 768, wt_pd_lin, pd_lin_b, logits, V, V, 768, total, 0);
    }
    log_softmax_k<<<(B * T) / 8, 256, 0, stream>>>(logits, out, B * T, V + MEL);

    // ---- mel head: 256->128, 2 GLU blocks at C=128, 128->80 tanh -> out[:, 256:336] ----
    {
        int total = (B * T / 16) * (DH / 16);
        gemm_bias<<<(total + 7) / 8, 256, 0, stream>>>(
            logits, V, wt_md_proj, md_proj_b, buf1, DH, DH, V, total, 0);
    }
    {
        int tilesT = T / 16;
        float* pin = buf1; float* pout = buf0;
        for (int i = 0; i < 2; ++i) {
            glu_conv_block<<<B * tilesT, 256, 0, stream>>>(
                pin, pout, wt_md + (size_t)i*(2*DH)*(3*DH), md_b + (size_t)i*2*DH,
                mel_lens, 0, zrow, T, T, DH, DH, tilesT);
            float* tw = pin; pin = pout; pout = tw;
        }   // result in buf1
    }
    {
        int total = (B * T / 16) * (MEL / 16);
        gemm_bias<<<(total + 7) / 8, 256, 0, stream>>>(
            buf1, DH, wt_md_lin, md_lin_b, out + V, V + MEL, MEL, DH, total, 1);
    }
    #undef GRID1
}
